// HierarchicalOPTFM_60610578481389
// MI455X (gfx1250) — matrix operand order repaired and checked by reading, not compiled, not run on hardware
//
#include <hip/hip_runtime.h>
#include <hip/hip_bf16.h>

// ---------------------------------------------------------------------------
// HierarchicalOPTFM forward for MI455X (gfx1250, wave32, WMMA).
//   * All H=64 GEMMs use v_wmma_f32_16x16x32_f16 (fp16 A/B, fp32 acc),
//     weights staged fp32->fp16 transposed in LDS, reused by all 8 waves.
//   * GEMM templated on <KIN, FLAGS>: KIN=64 path does unguarded b128 A loads,
//     epilogue branches resolved at compile time.
//   * Sparse E via edge gather/scatter-add (fp32 atomics); dense E never built.
//   * trans_conv: K^T V is a WMMA reduction GEMM over N; chunks staged into
//     LDS with gfx1250 async loads (global_load_async_to_lds_b128 +
//     s_wait_asynccnt), converted fp32->fp16 at fragment read.
// ---------------------------------------------------------------------------

typedef __attribute__((ext_vector_type(16))) _Float16 v16h;
typedef __attribute__((ext_vector_type(8)))  float    v8f;
typedef __attribute__((ext_vector_type(4)))  float    v4f;

#define NVAR 12288
#define NCON 6144
#define NEDG 196608

enum {
  GF_BIAS   = 1,   // add bias vector (64)
  GF_RELU   = 2,   // relu AFTER everything (incl accum)
  GF_ACCUM  = 4,   // Y += result (read-modify-write, non-atomic)
  GF_ADDSC  = 8,   // result += alpha * addSrc[row] for row < addLimit
  GF_ATOMIC = 16,  // atomicAdd into Y[scatterIdx[row]]
  GF_INRELU = 32,  // relu applied to A operand while loading
};

static __host__ __device__ __forceinline__ int imin(int a, int b) { return a < b ? a : b; }

// gfx1250 async global->LDS copy, 16 bytes per lane; tracked by ASYNCcnt.
static __device__ __forceinline__ void async_g2l_b128(unsigned lds_off, const float* gptr) {
  asm volatile("global_load_async_to_lds_b128 %0, %1, off"
               :: "v"(lds_off), "v"(gptr) : "memory");
}
static __device__ __forceinline__ void wait_asynccnt0() {
  asm volatile("s_wait_asynccnt 0x0" ::: "memory");
}

// ---------------------------------------------------------------------------
// Y[N x 64] = epilogue( X[N x KIN] @ W[KIN x 64] )  with optional row gather
// on X and atomic row scatter on Y. Block = 256 threads = 8 waves; each wave
// computes a 16-row x 64-col stripe with 4 WMMA tiles sharing one A fragment.
// ---------------------------------------------------------------------------
template <int KIN, int FLAGS>
__global__ void __launch_bounds__(256)
gemm64_wmma(const float* __restrict__ X, const int* __restrict__ gatherIdx,
            const float* __restrict__ W, const float* __restrict__ bias,
            float* __restrict__ Y, const int* __restrict__ scatterIdx,
            const float* __restrict__ addSrc, float addAlpha, int addLimit,
            int N)
{
  constexpr int KP = ((KIN + 31) / 32) * 32;   // padded K (32 or 64)
  __shared__ _Float16 WT[64 * KP];             // transposed weight: WT[col*KP + k]

  for (int i = threadIdx.x; i < 64 * KP; i += 256) {
    int c = i / KP, k = i - c * KP;
    WT[c * KP + k] = (k < KIN) ? (_Float16)W[(size_t)k * 64 + c] : (_Float16)0.f;
  }
  __syncthreads();

  const int wave = threadIdx.x >> 5;
  const int lane = threadIdx.x & 31;
  const int row0 = blockIdx.x * 128 + wave * 16;
  if (row0 >= N) return;                       // wave-uniform exit

  const int lh = lane & 15;
  const int hi = lane >> 4;

  // A fragment row for this lane (lanes l and l+16 share row l, split K)
  int arow = row0 + lh;
  int asrc = imin(arow, N - 1);
  if (gatherIdx) asrc = gatherIdx[asrc];
  const float* xrow = X + (size_t)asrc * KIN;
  __builtin_prefetch(xrow, 0, 3);

  v8f acc[4];
#pragma unroll
  for (int t = 0; t < 4; ++t) {
    float bv = (FLAGS & GF_BIAS) ? bias[t * 16 + lh] : 0.f;
    acc[t] = (v8f){bv, bv, bv, bv, bv, bv, bv, bv};
  }

  constexpr int ksteps = KP / 32;
#pragma unroll
  for (int ks = 0; ks < ksteps; ++ks) {
    const int k0 = ks * 32;
    // A layout: lane<16 -> K {0..7, 16..23}; lane>=16 -> K {8..15, 24..31}
    v16h a;
    const int kb = k0 + hi * 8;
    if constexpr (KIN % 32 == 0) {
      // unguarded, vectorized: 4 x b128 per k-step
      v4f f1a = *(const v4f*)(xrow + kb);
      v4f f1b = *(const v4f*)(xrow + kb + 4);
      v4f f2a = *(const v4f*)(xrow + kb + 16);
      v4f f2b = *(const v4f*)(xrow + kb + 20);
      if constexpr (FLAGS & GF_INRELU) {
#pragma unroll
        for (int j = 0; j < 4; ++j) {
          f1a[j] = fmaxf(f1a[j], 0.f); f1b[j] = fmaxf(f1b[j], 0.f);
          f2a[j] = fmaxf(f2a[j], 0.f); f2b[j] = fmaxf(f2b[j], 0.f);
        }
      }
#pragma unroll
      for (int j = 0; j < 4; ++j) {
        a[j]      = (_Float16)f1a[j];
        a[4 + j]  = (_Float16)f1b[j];
        a[8 + j]  = (_Float16)f2a[j];
        a[12 + j] = (_Float16)f2b[j];
      }
    } else {
#pragma unroll
      for (int j = 0; j < 8; ++j) {
        int k1 = kb + j, k2 = kb + 16 + j;
        float f1 = (k1 < KIN) ? xrow[k1] : 0.f;
        float f2 = (k2 < KIN) ? xrow[k2] : 0.f;
        a[j]     = (_Float16)f1;
        a[8 + j] = (_Float16)f2;
      }
    }
#pragma unroll
    for (int t = 0; t < 4; ++t) {
      // B layout: lane<16 -> col n0+lane, K {0..7, 16..23}; lane>=16 -> K {8..15, 24..31}
      const _Float16* bp = &WT[(t * 16 + lh) * KP + k0 + hi * 8];
      v16h b;
#pragma unroll
      for (int j = 0; j < 16; ++j) b[j] = bp[j < 8 ? j : j + 8];
      acc[t] = __builtin_amdgcn_wmma_f32_16x16x32_f16(
          false, a, false, b, (short)0, acc[t], false, false);
    }
  }

  // C layout: VGPR r -> row (r | r+8 by lane half), col = tile*16 + (lane&15)
#pragma unroll
  for (int t = 0; t < 4; ++t) {
    const int col = t * 16 + lh;
#pragma unroll
    for (int r = 0; r < 8; ++r) {
      const int row = row0 + r + hi * 8;
      if (row >= N) continue;
      float v = acc[t][r];
      if constexpr ((FLAGS & GF_ADDSC) != 0) {
        if (row < addLimit) v += addAlpha * addSrc[(size_t)row * 64 + col];
      }
      if constexpr ((FLAGS & GF_ACCUM) != 0) v += Y[(size_t)row * 64 + col];
      if constexpr ((FLAGS & GF_RELU) != 0)  v = fmaxf(v, 0.f);
      if constexpr ((FLAGS & GF_ATOMIC) != 0)
        atomicAdd(&Y[(size_t)scatterIdx[row] * 64 + col], v);
      else
        Y[(size_t)row * 64 + col] = v;
    }
  }
}

// ---------------------------------------------------------------------------
// kvs[64x64] += ks^T @ vs  (reduction over N rows), WMMA with LDS chunking.
// Chunks staged via gfx1250 async global->LDS loads (ASYNCcnt). N % 32 == 0.
// 8 waves: wave -> (m-stripe = wave>>1, n-half = wave&1), 2 tiles per wave.
// ---------------------------------------------------------------------------
__global__ void __launch_bounds__(256)
ktv_wmma(const float* __restrict__ ks, const float* __restrict__ vs,
         float* __restrict__ kvs, int N)
{
  __shared__ float ksS[32 * 64];   // row-major chunk [k][col], fp32
  __shared__ float vsS[32 * 64];
  const int lane = threadIdx.x & 31;
  const int wave = threadIdx.x >> 5;
  const int lh = lane & 15, hi = lane >> 4;
  const int m0 = (wave >> 1) * 16;

  // each thread async-copies 8 floats (2 x b128) of ks and of vs per chunk
  const unsigned ksOff = (unsigned)(uintptr_t)&ksS[threadIdx.x * 8];
  const unsigned vsOff = (unsigned)(uintptr_t)&vsS[threadIdx.x * 8];

  v8f acc[2];
  acc[0] = (v8f){0, 0, 0, 0, 0, 0, 0, 0};
  acc[1] = (v8f){0, 0, 0, 0, 0, 0, 0, 0};

  const int chunks = N >> 5;
  for (int ch = blockIdx.x; ch < chunks; ch += gridDim.x) {
    const size_t base = (size_t)ch * 32 * 64 + threadIdx.x * 8;
    __syncthreads();
    async_g2l_b128(ksOff,     ks + base);
    async_g2l_b128(ksOff + 16, ks + base + 4);
    async_g2l_b128(vsOff,     vs + base);
    async_g2l_b128(vsOff + 16, vs + base + 4);
    wait_asynccnt0();
    __syncthreads();

    // A[m][k] = ks[kbase+k][m0+m]; convert fp32->fp16 at read
    v16h a;
    const float* ap = &ksS[m0 + lh];
#pragma unroll
    for (int j = 0; j < 8; ++j) {
      a[j]     = (_Float16)ap[(hi * 8 + j) * 64];
      a[8 + j] = (_Float16)ap[(16 + hi * 8 + j) * 64];
    }
#pragma unroll
    for (int tp = 0; tp < 2; ++tp) {
      const int n0 = (wave & 1) * 32 + tp * 16;
      const float* bp = &vsS[n0 + lh];
      v16h b;
#pragma unroll
      for (int j = 0; j < 16; ++j) b[j] = (_Float16)bp[(hi * 8 + j + (j < 8 ? 0 : 8)) * 64];
      acc[tp] = __builtin_amdgcn_wmma_f32_16x16x32_f16(
          false, a, false, b, (short)0, acc[tp], false, false);
    }
  }

#pragma unroll
  for (int tp = 0; tp < 2; ++tp) {
    const int col = (wave & 1) * 32 + tp * 16 + lh;
#pragma unroll
    for (int r = 0; r < 8; ++r) {
      const int row = m0 + r + hi * 8;
      atomicAdd(&kvs[row * 64 + col], acc[tp][r]);
    }
  }
}

// --------------------------- small utility kernels -------------------------

__global__ void zero_f32(float* p, size_t n) {
  size_t i = (size_t)blockIdx.x * blockDim.x + threadIdx.x;
  if (i < n) p[i] = 0.f;
}

__global__ void combine_half(const float* a, const float* b, float* y, size_t n) {
  size_t i = (size_t)blockIdx.x * blockDim.x + threadIdx.x;
  if (i < n) y[i] = 0.5f * a[i] + 0.5f * b[i];
}

__global__ void sumsq_f32(const float* __restrict__ x, float* out, size_t n) {
  float s = 0.f;
  for (size_t i = (size_t)blockIdx.x * blockDim.x + threadIdx.x; i < n;
       i += (size_t)gridDim.x * blockDim.x)
    s += x[i] * x[i];
  for (int o = 16; o; o >>= 1) s += __shfl_xor(s, o);
  if ((threadIdx.x & 31) == 0) atomicAdd(out, s);
}

__global__ void scale_invnorm(float* x, const float* ss, size_t n) {
  float inv = 1.f / (sqrtf(*ss) + 1e-8f);
  size_t i = (size_t)blockIdx.x * blockDim.x + threadIdx.x;
  if (i < n) x[i] *= inv;
}

__global__ void colsum64(const float* __restrict__ x, float* sum, int N) {
  float s = 0.f;
  int col = threadIdx.x & 63, rg = threadIdx.x >> 6;
  for (int r = blockIdx.x * 4 + rg; r < N; r += gridDim.x * 4)
    s += x[(size_t)r * 64 + col];
  atomicAdd(&sum[col], s);
}

// rows of 64; one wave per row. Optional pre-average with Xb, LN, optional relu.
__global__ void ln_kernel(const float* __restrict__ Xa, const float* __restrict__ Xb,
                          const float* __restrict__ g, const float* __restrict__ bt,
                          float* __restrict__ Y, int N, int doRelu)
{
  int row = blockIdx.x * 8 + (threadIdx.x >> 5);
  int lane = threadIdx.x & 31;
  if (row >= N) return;
  const float* ar = Xa + (size_t)row * 64;
  float x0 = ar[lane], x1 = ar[lane + 32];
  if (Xb) {
    const float* br = Xb + (size_t)row * 64;
    x0 = 0.5f * (x0 + br[lane]);
    x1 = 0.5f * (x1 + br[lane + 32]);
  }
  float s = x0 + x1;
  for (int o = 16; o; o >>= 1) s += __shfl_xor(s, o);
  float mu = s * (1.f / 64.f);
  float d0 = x0 - mu, d1 = x1 - mu;
  float q = d0 * d0 + d1 * d1;
  for (int o = 16; o; o >>= 1) q += __shfl_xor(q, o);
  float inv = rsqrtf(q * (1.f / 64.f) + 1e-5f);
  float y0 = d0 * inv * g[lane] + bt[lane];
  float y1 = d1 * inv * g[lane + 32] + bt[lane + 32];
  if (doRelu) { y0 = fmaxf(y0, 0.f); y1 = fmaxf(y1, 0.f); }
  Y[(size_t)row * 64 + lane] = y0;
  Y[(size_t)row * 64 + lane + 32] = y1;
}

// out[row] = num[row] / (dot(qs[row], ksum) + Nq); one wave per row, in-place ok
__global__ void conv_finalize(const float* __restrict__ num, const float* __restrict__ qs,
                              const float* __restrict__ ksum, float Nq,
                              float* __restrict__ Y, int N)
{
  int row = blockIdx.x * 8 + (threadIdx.x >> 5);
  int lane = threadIdx.x & 31;
  if (row >= N) return;
  const float* qr = qs + (size_t)row * 64;
  float d = qr[lane] * ksum[lane] + qr[lane + 32] * ksum[lane + 32];
  for (int o = 16; o; o >>= 1) d += __shfl_xor(d, o);
  float inv = 1.f / (d + Nq);
  Y[(size_t)row * 64 + lane]      = num[(size_t)row * 64 + lane] * inv;
  Y[(size_t)row * 64 + lane + 32] = num[(size_t)row * 64 + lane + 32] * inv;
}

// per-edge: T[e] = relu(L[li[e]] + relu(ef[e]*eW+eB) * feW + R[ri[e]]); wave/edge
__global__ void edge_msg(const float* __restrict__ L, const float* __restrict__ R,
                         const int* __restrict__ li, const int* __restrict__ ri,
                         const float* __restrict__ ef, const float* __restrict__ eW,
                         const float* __restrict__ eB, const float* __restrict__ feW,
                         float* __restrict__ T, int NE)
{
  int e = blockIdx.x * 8 + (threadIdx.x >> 5);
  int lane = threadIdx.x & 31;
  if (e >= NE) return;
  float es = fmaxf(ef[e] * eW[0] + eB[0], 0.f);
  const float* lr = L + (size_t)li[e] * 64;
  const float* rr = R + (size_t)ri[e] * 64;
  float t0 = lr[lane]      + es * feW[lane]      + rr[lane];
  float t1 = lr[lane + 32] + es * feW[lane + 32] + rr[lane + 32];
  T[(size_t)e * 64 + lane]      = fmaxf(t0, 0.f);
  T[(size_t)e * 64 + lane + 32] = fmaxf(t1, 0.f);
}

// dst[didx[e]] += src[sidx[e]] (64 floats); sparse E matvec via edges
__global__ void edge_scatter(float* __restrict__ dst, const float* __restrict__ src,
                             const int* __restrict__ sidx, const int* __restrict__ didx,
                             int NE)
{
  int e = blockIdx.x * 8 + (threadIdx.x >> 5);
  int lane = threadIdx.x & 31;
  if (e >= NE) return;
  const float* sr = src + (size_t)sidx[e] * 64;
  float* dr = dst + (size_t)didx[e] * 64;
  atomicAdd(&dr[lane], sr[lane]);
  atomicAdd(&dr[lane + 32], sr[lane + 32]);
}

// out[row][0:2] = [xv | exc] @ W(128x2) + b; one wave per row
__global__ void final_fc(const float* __restrict__ xv, const float* __restrict__ exc,
                         const float* __restrict__ W, const float* __restrict__ b,
                         float* __restrict__ out, int N)
{
  int row = blockIdx.x * 8 + (threadIdx.x >> 5);
  int lane = threadIdx.x & 31;
  if (row >= N) return;
  const float* xr = xv + (size_t)row * 64;
  const float* er = exc + (size_t)row * 64;
  float s0 = xr[lane] * W[lane * 2]            + xr[lane + 32] * W[(lane + 32) * 2]
           + er[lane] * W[(64 + lane) * 2]     + er[lane + 32] * W[(96 + lane) * 2];
  float s1 = xr[lane] * W[lane * 2 + 1]        + xr[lane + 32] * W[(lane + 32) * 2 + 1]
           + er[lane] * W[(64 + lane) * 2 + 1] + er[lane + 32] * W[(96 + lane) * 2 + 1];
  for (int o = 16; o; o >>= 1) { s0 += __shfl_xor(s0, o); s1 += __shfl_xor(s1, o); }
  if (lane == 0) {
    out[(size_t)row * 2]     = s0 + b[0];
    out[(size_t)row * 2 + 1] = s1 + b[1];
  }
}

// ---------------------------------------------------------------------------

extern "C" void kernel_launch(void* const* d_in, const int* in_sizes, int n_in,
                              void* d_out, int out_size, void* d_ws, size_t ws_size,
                              hipStream_t stream) {
  (void)in_sizes; (void)n_in; (void)out_size; (void)ws_size;

  auto P = [&](int i) { return (const float*)d_in[i]; };
  const float* vfeat = P(0);
  const float* cfeat = P(1);
  const float* efeat = P(2);
  const int* eidx = (const int*)d_in[3];
  const int* eCons = eidx;          // edge_indices[0]
  const int* eVar  = eidx + NEDG;   // edge_indices[1]

  // workspace carve-up (fp32)
  float* w = (float*)d_ws;
  size_t off = 0;
  auto alloc = [&](size_t n) { float* p = w + off; off += n; return p; };
  const size_t NVH = (size_t)NVAR * 64, NCH = (size_t)NCON * 64, NEH = (size_t)NEDG * 64;
  float* pv[8]; for (int i = 0; i < 8; ++i) pv[i] = alloc(NVH);
  float* pc[8]; for (int i = 0; i < 8; ++i) pc[i] = alloc(NCH);
  float* Tedge = alloc(NEH);
  float* kvs = alloc(64 * 64);
  float* ksum = alloc(64);
  float* ss = alloc(4);

  auto zero = [&](float* p, size_t n) {
    zero_f32<<<dim3((unsigned)((n + 255) / 256)), 256, 0, stream>>>(p, n);
  };
  // dispatch over compile-time <KIN, FLAGS>
  auto gemm = [&](const float* X, const int* gi, const float* W, const float* b,
                  float* Y, const int* si, const float* addS, float alpha, int alim,
                  int N, int KIN, int flags) {
    dim3 grid((N + 127) / 128), blk(256);
    if (KIN == 9) {
      gemm64_wmma<9, GF_BIAS><<<grid, blk, 0, stream>>>(X, gi, W, b, Y, si, addS, alpha, alim, N);
      return;
    }
    if (KIN == 1) {
      gemm64_wmma<1, GF_BIAS><<<grid, blk, 0, stream>>>(X, gi, W, b, Y, si, addS, alpha, alim, N);
      return;
    }
    switch (flags) {
      case 0:
        gemm64_wmma<64, 0><<<grid, blk, 0, stream>>>(X, gi, W, b, Y, si, addS, alpha, alim, N); break;
      case GF_BIAS:
        gemm64_wmma<64, GF_BIAS><<<grid, blk, 0, stream>>>(X, gi, W, b, Y, si, addS, alpha, alim, N); break;
      case GF_BIAS | GF_RELU:
        gemm64_wmma<64, GF_BIAS | GF_RELU><<<grid, blk, 0, stream>>>(X, gi, W, b, Y, si, addS, alpha, alim, N); break;
      case GF_ACCUM:
        gemm64_wmma<64, GF_ACCUM><<<grid, blk, 0, stream>>>(X, gi, W, b, Y, si, addS, alpha, alim, N); break;
      case GF_ACCUM | GF_RELU:
        gemm64_wmma<64, GF_ACCUM | GF_RELU><<<grid, blk, 0, stream>>>(X, gi, W, b, Y, si, addS, alpha, alim, N); break;
      case GF_ADDSC:
        gemm64_wmma<64, GF_ADDSC><<<grid, blk, 0, stream>>>(X, gi, W, b, Y, si, addS, alpha, alim, N); break;
      case GF_BIAS | GF_ATOMIC:
        gemm64_wmma<64, GF_BIAS | GF_ATOMIC><<<grid, blk, 0, stream>>>(X, gi, W, b, Y, si, addS, alpha, alim, N); break;
      case GF_BIAS | GF_INRELU:
        gemm64_wmma<64, GF_BIAS | GF_INRELU><<<grid, blk, 0, stream>>>(X, gi, W, b, Y, si, addS, alpha, alim, N); break;
      default:
        gemm64_wmma<64, GF_BIAS><<<grid, blk, 0, stream>>>(X, gi, W, b, Y, si, addS, alpha, alim, N); break;
    }
  };
  auto ln = [&](const float* a, const float* b2, const float* g, const float* bb,
                float* y, int N, int relu) {
    ln_kernel<<<dim3((N + 7) / 8), 256, 0, stream>>>(a, b2, g, bb, y, N, relu);
  };
  auto transconv = [&](const float* qin, int Nq, const float* sin, int Ns,
                       const float* WkW, const float* Wkb, const float* WqW,
                       const float* Wqb, const float* WvW, const float* Wvb,
                       float* qs, float* ks, float* vs, float* out) {
    gemm(qin, nullptr, WqW, Wqb, qs, nullptr, nullptr, 0.f, 0, Nq, 64, GF_BIAS);
    gemm(sin, nullptr, WkW, Wkb, ks, nullptr, nullptr, 0.f, 0, Ns, 64, GF_BIAS);
    gemm(sin, nullptr, WvW, Wvb, vs, nullptr, nullptr, 0.f, 0, Ns, 64, GF_BIAS);
    zero(ss, 2);
    sumsq_f32<<<dim3(1024), 256, 0, stream>>>(qs, ss + 0, (size_t)Nq * 64);
    sumsq_f32<<<dim3(1024), 256, 0, stream>>>(ks, ss + 1, (size_t)Ns * 64);
    scale_invnorm<<<dim3((unsigned)(((size_t)Nq * 64 + 255) / 256)), 256, 0, stream>>>(qs, ss + 0, (size_t)Nq * 64);
    scale_invnorm<<<dim3((unsigned)(((size_t)Ns * 64 + 255) / 256)), 256, 0, stream>>>(ks, ss + 1, (size_t)Ns * 64);
    zero(kvs, 64 * 64);
    zero(ksum, 64);
    ktv_wmma<<<dim3(256), 256, 0, stream>>>(ks, vs, kvs, Ns);
    colsum64<<<dim3(512), 256, 0, stream>>>(ks, ksum, Ns);
    int alim = imin(Nq, Ns);
    // num = qs @ kvs + Nq * vs (rows < alim)
    gemm(qs, nullptr, kvs, nullptr, out, nullptr, vs, (float)Nq, alim, Nq, 64, GF_ADDSC);
    conv_finalize<<<dim3((Nq + 7) / 8), 256, 0, stream>>>(out, qs, ksum, (float)Nq, out, Nq);
  };

  // ---- embeddings -----------------------------------------------------
  gemm(vfeat, nullptr, P(98), P(99), pv[0], nullptr, nullptr, 0.f, 0, NVAR, 9, GF_BIAS);
  gemm(cfeat, nullptr, P(4),  P(5),  pc[0], nullptr, nullptr, 0.f, 0, NCON, 1, GF_BIAS);

  // ---- self_var (params 86..97) -> v1 in pv[6] ------------------------
  gemm(pv[0], nullptr, P(96), P(97), pv[1], nullptr, nullptr, 0.f, 0, NVAR, 64, GF_BIAS);
  ln(pv[1], nullptr, P(87), P(86), pv[1], NVAR, 1);
  transconv(pv[1], NVAR, pv[1], NVAR, P(90), P(91), P(92), P(93), P(94), P(95),
            pv[2], pv[3], pv[4], pv[5]);
  ln(pv[5], pv[1], P(89), P(88), pv[6], NVAR, 1);

  // ---- self_cons (params 74..85) -> c1 in pc[1] -----------------------
  gemm(pc[0], nullptr, P(84), P(85), pc[2], nullptr, nullptr, 0.f, 0, NCON, 64, GF_BIAS);
  ln(pc[2], nullptr, P(75), P(74), pc[2], NCON, 1);
  transconv(pc[2], NCON, pc[2], NCON, P(78), P(79), P(80), P(81), P(82), P(83),
            pc[3], pc[4], pc[5], pc[6]);
  ln(pc[6], pc[2], P(77), P(76), pc[1], NCON, 1);

  // ---- cross_cv (params 6..19): x=v1, y=c1 -> v2 in pv[7] -------------
  ln(pv[6], nullptr, P(7), P(6), pv[1], NVAR, 1);       // xb
  ln(pc[1], nullptr, P(9), P(8), pc[2], NCON, 1);       // yb
  transconv(pv[1], NVAR, pc[2], NCON, P(12), P(13), P(14), P(15), P(16), P(17),
            pv[2], pc[3], pc[4], pv[3]);
  ln(pv[3], pv[1], P(11), P(10), pv[4], NVAR, 1);       // x3
  zero(pv[5], NVH);
  edge_scatter<<<dim3(NEDG / 8), 256, 0, stream>>>(pv[5], pc[2], eCons, eVar, NEDG); // E^T yb
  gemm(pv[4], nullptr, P(18), P(19), pv[7], nullptr, nullptr, 0.f, 0, NVAR, 64, GF_BIAS);
  gemm(pv[5], nullptr, P(18) + 64 * 64, nullptr, pv[7], nullptr, nullptr, 0.f, 0, NVAR, 64, GF_ACCUM);

  // ---- cross_vc (params 20..33): x=c1, y=v1 -> c2 in pc[7] ------------
  ln(pc[1], nullptr, P(21), P(20), pc[2], NCON, 1);     // xb
  ln(pv[6], nullptr, P(23), P(22), pv[1], NVAR, 1);     // yb
  transconv(pc[2], NCON, pv[1], NVAR, P(26), P(27), P(28), P(29), P(30), P(31),
            pc[3], pv[2], pv[3], pc[4]);
  ln(pc[4], pc[2], P(25), P(24), pc[5], NCON, 1);       // x3
  zero(pc[6], NCH);
  edge_scatter<<<dim3(NEDG / 8), 256, 0, stream>>>(pc[6], pv[1], eVar, eCons, NEDG); // E yb
  gemm(pc[5], nullptr, P(32), P(33), pc[7], nullptr, nullptr, 0.f, 0, NCON, 64, GF_BIAS);
  gemm(pc[6], nullptr, P(32) + 64 * 64, nullptr, pc[7], nullptr, nullptr, 0.f, 0, NCON, 64, GF_ACCUM);

  // ---- GNN ------------------------------------------------------------
  // embeddings seq2
  gemm(pc[0], nullptr, P(48), P(49), pc[1], nullptr, nullptr, 0.f, 0, NCON, 64, GF_BIAS | GF_RELU);
  gemm(pc[1], nullptr, P(50), P(51), pc[2], nullptr, nullptr, 0.f, 0, NCON, 64, GF_BIAS | GF_RELU); // c
  gemm(pv[0], nullptr, P(70), P(71), pv[1], nullptr, nullptr, 0.f, 0, NVAR, 64, GF_BIAS | GF_RELU);
  gemm(pv[1], nullptr, P(72), P(73), pv[2], nullptr, nullptr, 0.f, 0, NVAR, 64, GF_BIAS | GF_RELU); // v

  // v2c (params 58..69): left=v, right=c, li=eVar, ri=eCons
  gemm(pv[2], nullptr, P(61), P(62), pv[3], nullptr, nullptr, 0.f, 0, NVAR, 64, GF_BIAS);  // Lv
  gemm(pc[2], nullptr, P(63), nullptr, pc[3], nullptr, nullptr, 0.f, 0, NCON, 64, 0);      // Rc
  edge_msg<<<dim3(NEDG / 8), 256, 0, stream>>>(pv[3], pc[3], eVar, eCons, efeat,
                                               P(52), P(53), P(58), Tedge, NEDG);
  zero(pc[4], NCH);
  gemm(Tedge, nullptr, P(59), P(60), pc[4], eCons, nullptr, 0.f, 0, NEDG, 64, GF_BIAS | GF_ATOMIC);
  gemm(pc[4], nullptr, P(68), P(69), pc[5], nullptr, nullptr, 0.f, 0, NCON, 64, GF_BIAS | GF_INRELU); // h
  gemm(pc[5], nullptr, P(64), P(65), pc[6], nullptr, nullptr, 0.f, 0, NCON, 64, GF_BIAS);
  gemm(pc[2], nullptr, P(64) + 64 * 64, nullptr, pc[6], nullptr, nullptr, 0.f, 0, NCON, 64, GF_ACCUM | GF_RELU);
  gemm(pc[6], nullptr, P(66), P(67), pc[1], nullptr, nullptr, 0.f, 0, NCON, 64, GF_BIAS); // c_new

  // c2v (params 36..47): left=c_new, right=v, li=eCons, ri=eVar
  gemm(pc[1], nullptr, P(39), P(40), pc[3], nullptr, nullptr, 0.f, 0, NCON, 64, GF_BIAS);  // Lc
  gemm(pv[2], nullptr, P(41), nullptr, pv[4], nullptr, nullptr, 0.f, 0, NVAR, 64, 0);      // Rv
  edge_msg<<<dim3(NEDG / 8), 256, 0, stream>>>(pc[3], pv[4], eCons, eVar, efeat,
                                               P(52), P(53), P(36), Tedge, NEDG);
  zero(pv[5], NVH);
  gemm(Tedge, nullptr, P(37), P(38), pv[5], eVar, nullptr, 0.f, 0, NEDG, 64, GF_BIAS | GF_ATOMIC);
  gemm(pv[5], nullptr, P(46), P(47), pv[1], nullptr, nullptr, 0.f, 0, NVAR, 64, GF_BIAS | GF_INRELU); // h
  gemm(pv[1], nullptr, P(42), P(43), pv[3], nullptr, nullptr, 0.f, 0, NVAR, 64, GF_BIAS);
  gemm(pv[2], nullptr, P(42) + 64 * 64, nullptr, pv[3], nullptr, nullptr, 0.f, 0, NVAR, 64, GF_ACCUM | GF_RELU);
  gemm(pv[3], nullptr, P(44), P(45), pv[4], nullptr, nullptr, 0.f, 0, NVAR, 64, GF_BIAS); // v_new

  gemm(pv[4], nullptr, P(56), P(57), pv[1], nullptr, nullptr, 0.f, 0, NVAR, 64, GF_BIAS | GF_RELU); // vg
  gemm(pc[1], nullptr, P(54), P(55), pc[3], nullptr, nullptr, 0.f, 0, NCON, 64, GF_BIAS | GF_RELU); // cg

  // ---- combine + final fc --------------------------------------------
  combine_half<<<dim3((unsigned)((NVH + 255) / 256)), 256, 0, stream>>>(pv[1], pv[7], pv[2], NVH); // xv
  combine_half<<<dim3((unsigned)((NCH + 255) / 256)), 256, 0, stream>>>(pc[3], pc[7], pc[2], NCH); // xc
  zero(pv[3], NVH);
  edge_scatter<<<dim3(NEDG / 8), 256, 0, stream>>>(pv[3], pc[2], eCons, eVar, NEDG); // E^T xc
  final_fc<<<dim3((NVAR + 7) / 8), 256, 0, stream>>>(pv[2], pv[3], P(34), P(35),
                                                     (float*)d_out, NVAR);
}
